// BasicLSTM_33517924778111
// MI455X (gfx1250) — matrix operand order repaired and checked by reading, not compiled, not run on hardware
//
#include <hip/hip_runtime.h>
#include <hip/hip_bf16.h>
#include <math.h>

// ---------------------------------------------------------------------------
// Fused 2-layer LSTM (H=64, S=512, B=2048) + final FC(64->12), CDNA5 WMMA.
//
//  * grid = B/16 = 128 workgroups, 256 threads (8 wave32s) each; each WG owns
//    16 batch rows for the whole sequence. Zero global intermediate traffic.
//  * Per timestep (4 barriers):
//      A: gates0 = h1 @ w_hh0^T                (4 WMMA/wave, bias in C operand)
//      B: layer0 pointwise (v_tanh_f32)  -> h1_t
//      C: gates1 = h1_t @ w_ih1^T + h2 @ w_hh1^T  (8 WMMA/wave)
//      D: layer1 pointwise               -> h2
//  * Weight B-fragments register-resident (12 x v16h = 96 VGPRs, loaded once).
//  * h-state in LDS as f16 with 72-half row pitch -> A fragments are 2x
//    ds_load_b128, conflict-balanced. Gate staging via LDS float4 ops.
//  * Gate biases pre-splatted into v8f WMMA C-init vectors (column == lane).
// ---------------------------------------------------------------------------

typedef __attribute__((ext_vector_type(16))) _Float16 v16h;
typedef __attribute__((ext_vector_type(8)))  float    v8f;

#define LSTM_H   64
#define LSTM_G   256     // 4*H
#define LSTM_S   512
#define ROWS     16      // batch rows per workgroup
#define HPAD     72      // h-state row pitch in halves (144B: 16B-aligned)

union V16H {
    v16h v;
    unsigned int u[8];
    _Float16 h[16];
};
union F4 { float4 v; float f[4]; };

// A fragment (16x32 f16, M=batch rows): lane L -> m = L%16, half = L/16.
// Lane holds 16 contiguous halves: K = kbase+16*half+{0..15} (matches B frag).
__device__ __forceinline__ v16h load_afrag(const _Float16* hlds, int m, int half,
                                           int kbase) {
    const _Float16* p = hlds + m * HPAD + kbase + 16 * half;
    uint4 lo = *(const uint4*)(p);        // 16B aligned: 144*m + 16*(...)
    uint4 hi = *(const uint4*)(p + 8);
    V16H r;
    r.u[0] = lo.x; r.u[1] = lo.y; r.u[2] = lo.z; r.u[3] = lo.w;
    r.u[4] = hi.x; r.u[5] = hi.y; r.u[6] = hi.z; r.u[7] = hi.w;
    return r.v;
}

// B fragment (32x16 f16): B[k][n] = W[gbase+n][kbase+k], W row-major [256][64].
// lane L -> n = L%16, khalf = L/16; element j holds K = 16*khalf + j, i.e. 16
// consecutive floats of one weight row -> 4x global float4 loads + cvt.
__device__ __forceinline__ v16h load_bfrag(const float* W, int gbase, int n,
                                           int khalf, int kbase) {
    const float* src = W + (gbase + n) * LSTM_H + kbase + 16 * khalf;
    V16H r;
    #pragma unroll
    for (int q = 0; q < 4; ++q) {
        float4 f = *(const float4*)(src + 4 * q);
        r.h[4*q + 0] = (_Float16)f.x;
        r.h[4*q + 1] = (_Float16)f.y;
        r.h[4*q + 2] = (_Float16)f.z;
        r.h[4*q + 3] = (_Float16)f.w;
    }
    return r.v;
}

// CDNA5 hardware tanh (TRANS op). Trailing v_nop covers the documented
// TRANS->consumer RAW hazard that the compiler cannot see through inline asm.
__device__ __forceinline__ float tanh_hw(float x) {
    float y;
    asm("v_tanh_f32 %0, %1\n\tv_nop" : "=v"(y) : "v"(x));
    return y;
}
__device__ __forceinline__ float sigmoid_hw(float x) {
    return fmaf(0.5f, tanh_hw(0.5f * x), 0.5f);
}

#define WMMA_F16(A, B, C) \
    __builtin_amdgcn_wmma_f32_16x16x32_f16(false, (A), false, (B), (short)0, \
                                           (C), false, false)

__global__ __launch_bounds__(256) void lstm2_fused_kernel(
    const float* __restrict__ xg,      // [B, S, 1]
    const float* __restrict__ w_ih0,   // [256, 1]
    const float* __restrict__ w_hh0,   // [256, 64]
    const float* __restrict__ b_ih0,   // [256]
    const float* __restrict__ b_hh0,   // [256]
    const float* __restrict__ w_ih1,   // [256, 64]
    const float* __restrict__ w_hh1,   // [256, 64]
    const float* __restrict__ b_ih1,   // [256]
    const float* __restrict__ b_hh1,   // [256]
    const float* __restrict__ fc_w,    // [12, 64]
    const float* __restrict__ fc_b,    // [12]
    float* __restrict__ out)           // [B, 12]
{
    __shared__ __align__(16) float    xAll[ROWS * LSTM_S];    // 32 KB x tile
    __shared__ __align__(16) float    gatesS[ROWS * LSTM_G];  // 16 KB gates
    __shared__ __align__(16) _Float16 hA[ROWS * HPAD];        // h1 state (f16)
    __shared__ __align__(16) _Float16 hB[ROWS * HPAD];        // h2 state (f16)
    __shared__ __align__(16) float    hfin[ROWS * LSTM_H];    // final h2 (f32)

    const int tid   = threadIdx.x;
    const int lane  = tid & 31;
    const int wave  = tid >> 5;
    const int nlane = lane & 15;   // N within tile (B,C,D) / M for A frags
    const int khalf = lane >> 4;   // lane-half selector
    const int gb0   = blockIdx.x * ROWS;

    // elementwise mapping: row em, 4 consecutive units eu0..eu0+3 per thread
    const int em  = tid >> 4;
    const int eu0 = (tid & 15) << 2;

    // ---- one-time setup -------------------------------------------------
    for (int i = tid; i < ROWS * HPAD; i += 256) {
        hA[i] = (_Float16)0.0f;
        hB[i] = (_Float16)0.0f;
    }
    {   // coalesced flat copy of this block's x tile
        const float* xrow = xg + (size_t)gb0 * LSTM_S;
        for (int i = tid; i < ROWS * LSTM_S; i += 256) xAll[i] = xrow[i];
    }

    // Register-resident weight fragments (loaded once).
    v16h Brec0[2][2], Bih1[2][2], Bhh1[2][2];
    // Per-gate-column bias splats used as WMMA C-init (column == lane in C/D).
    v8f cinit0[2], cinit1[2];
    #pragma unroll
    for (int nt = 0; nt < 2; ++nt) {
        const int gbase = (wave * 2 + nt) * 16;
        #pragma unroll
        for (int kf = 0; kf < 2; ++kf) {
            Brec0[nt][kf] = load_bfrag(w_hh0, gbase, nlane, khalf, kf * 32);
            Bih1[nt][kf]  = load_bfrag(w_ih1, gbase, nlane, khalf, kf * 32);
            Bhh1[nt][kf]  = load_bfrag(w_hh1, gbase, nlane, khalf, kf * 32);
        }
        const int col = gbase + nlane;
        const float bb0 = b_ih0[col] + b_hh0[col];
        const float bb1 = b_ih1[col] + b_hh1[col];
        #pragma unroll
        for (int r = 0; r < 8; ++r) { cinit0[nt][r] = bb0; cinit1[nt][r] = bb1; }
    }

    // w_ih0 factors for this thread's 4 units x 4 gates (input size == 1).
    float wreg[16];
    #pragma unroll
    for (int q = 0; q < 4; ++q)
        #pragma unroll
        for (int j = 0; j < 4; ++j)
            wreg[q * 4 + j] = w_ih0[q * 64 + eu0 + j];

    float c0[4] = {0.f, 0.f, 0.f, 0.f};
    float c1[4] = {0.f, 0.f, 0.f, 0.f};

    __syncthreads();

    // ---- recurrence -----------------------------------------------------
    for (int t = 0; t < LSTM_S; ++t) {
        // phase A: gates0 = h1 @ w_hh0^T + bias0
        {
            v16h a0 = load_afrag(hA, nlane, khalf, 0);
            v16h a1 = load_afrag(hA, nlane, khalf, 32);
            #pragma unroll
            for (int nt = 0; nt < 2; ++nt) {
                v8f acc = WMMA_F16(a0, Brec0[nt][0], cinit0[nt]);
                acc     = WMMA_F16(a1, Brec0[nt][1], acc);
                const int col = (wave * 2 + nt) * 16 + nlane;
                #pragma unroll
                for (int r = 0; r < 8; ++r)
                    gatesS[(r + 8 * khalf) * LSTM_G + col] = acc[r];
            }
        }
        __syncthreads();

        // phase B: layer0 pointwise -> h1_t (into hA)
        {
            const float xv = xAll[em * LSTM_S + t];
            F4 ai, af, ag, ao;
            ai.v = *(const float4*)&gatesS[em * LSTM_G + eu0      ];
            af.v = *(const float4*)&gatesS[em * LSTM_G + eu0 +  64];
            ag.v = *(const float4*)&gatesS[em * LSTM_G + eu0 + 128];
            ao.v = *(const float4*)&gatesS[em * LSTM_G + eu0 + 192];
            union { _Float16 h4[4]; uint2 u2; } pk;
            #pragma unroll
            for (int j = 0; j < 4; ++j) {
                const float gi = fmaf(xv, wreg[0 * 4 + j], ai.f[j]);
                const float gf = fmaf(xv, wreg[1 * 4 + j], af.f[j]);
                const float gg = fmaf(xv, wreg[2 * 4 + j], ag.f[j]);
                const float go = fmaf(xv, wreg[3 * 4 + j], ao.f[j]);
                const float c = sigmoid_hw(gf) * c0[j]
                              + sigmoid_hw(gi) * tanh_hw(gg);
                c0[j] = c;
                pk.h4[j] = (_Float16)(sigmoid_hw(go) * tanh_hw(c));
            }
            *(uint2*)&hA[em * HPAD + eu0] = pk.u2;   // 8B-aligned b64 store
        }
        __syncthreads();

        // phase C: gates1 = h1_t @ w_ih1^T + h2 @ w_hh1^T + bias1
        {
            v16h p0 = load_afrag(hA, nlane, khalf, 0);
            v16h p1 = load_afrag(hA, nlane, khalf, 32);
            v16h q0 = load_afrag(hB, nlane, khalf, 0);
            v16h q1 = load_afrag(hB, nlane, khalf, 32);
            #pragma unroll
            for (int nt = 0; nt < 2; ++nt) {
                v8f acc = WMMA_F16(q0, Bhh1[nt][0], cinit1[nt]);
                acc     = WMMA_F16(q1, Bhh1[nt][1], acc);
                acc     = WMMA_F16(p0, Bih1[nt][0], acc);
                acc     = WMMA_F16(p1, Bih1[nt][1], acc);
                const int col = (wave * 2 + nt) * 16 + nlane;
                #pragma unroll
                for (int r = 0; r < 8; ++r)
                    gatesS[(r + 8 * khalf) * LSTM_G + col] = acc[r];
            }
        }
        __syncthreads();

        // phase D: layer1 pointwise -> h2 (into hB)
        {
            F4 ai, af, ag, ao, hv;
            ai.v = *(const float4*)&gatesS[em * LSTM_G + eu0      ];
            af.v = *(const float4*)&gatesS[em * LSTM_G + eu0 +  64];
            ag.v = *(const float4*)&gatesS[em * LSTM_G + eu0 + 128];
            ao.v = *(const float4*)&gatesS[em * LSTM_G + eu0 + 192];
            union { _Float16 h4[4]; uint2 u2; } pk;
            #pragma unroll
            for (int j = 0; j < 4; ++j) {
                const float c = sigmoid_hw(af.f[j]) * c1[j]
                              + sigmoid_hw(ai.f[j]) * tanh_hw(ag.f[j]);
                c1[j] = c;
                const float h = sigmoid_hw(ao.f[j]) * tanh_hw(c);
                hv.f[j] = h;
                pk.h4[j] = (_Float16)h;
            }
            *(uint2*)&hB[em * HPAD + eu0] = pk.u2;
            if (t == LSTM_S - 1)                       // uniform branch
                *(float4*)&hfin[em * LSTM_H + eu0] = hv.v;
        }
        __syncthreads();
    }

    // ---- final FC: out[b] = h2_last @ fc_w^T + fc_b ---------------------
    if (tid < ROWS * 12) {
        const int m = tid / 12, o = tid % 12;
        float s = fc_b[o];
        #pragma unroll
        for (int u = 0; u < LSTM_H; ++u)
            s += hfin[m * LSTM_H + u] * fc_w[o * LSTM_H + u];
        out[(size_t)(gb0 + m) * 12 + o] = s;
    }
}

extern "C" void kernel_launch(void* const* d_in, const int* in_sizes, int n_in,
                              void* d_out, int out_size, void* d_ws, size_t ws_size,
                              hipStream_t stream) {
    (void)in_sizes; (void)n_in; (void)out_size; (void)d_ws; (void)ws_size;
    const float* x     = (const float*)d_in[0];
    const float* w_ih0 = (const float*)d_in[1];
    const float* w_hh0 = (const float*)d_in[2];
    const float* b_ih0 = (const float*)d_in[3];
    const float* b_hh0 = (const float*)d_in[4];
    const float* w_ih1 = (const float*)d_in[5];
    const float* w_hh1 = (const float*)d_in[6];
    const float* b_ih1 = (const float*)d_in[7];
    const float* b_hh1 = (const float*)d_in[8];
    const float* fc_w  = (const float*)d_in[9];
    const float* fc_b  = (const float*)d_in[10];
    float* out = (float*)d_out;

    dim3 grid(2048 / ROWS);   // 128 workgroups, one per 16 batch rows
    dim3 block(256);          // 8 wave32s
    lstm2_fused_kernel<<<grid, block, 0, stream>>>(
        x, w_ih0, w_hh0, b_ih0, b_hh0, w_ih1, w_hh1, b_ih1, b_hh1,
        fc_w, fc_b, out);
}
